// GPUAcceleratedLTM_56985626083904
// MI455X (gfx1250) — matrix operand order repaired and checked by reading, not compiled, not run on hardware
//
#include <hip/hip_runtime.h>
#include <hip/hip_fp16.h>
#include <math.h>

typedef __attribute__((ext_vector_type(16))) _Float16 v16h;
typedef __attribute__((ext_vector_type(8)))  _Float16 v8h;
typedef __attribute__((ext_vector_type(8)))  float    v8f;

#define NB   1024      // batch (queries)
#define ND   1024      // full dim
#define NC   256       // compressed dim
#define NKEY 100000    // stored keys
#define TOPK 5
#define SLICES 8
#define KEYS_PER_SLICE (NKEY / SLICES)   // 12500
#define QT   32        // queries per distance block

// ---------------- device helpers ----------------

__device__ __forceinline__ float gelu_exact(float x) {
    return 0.5f * x * (1.0f + erff(x * 0.70710678118654752f));
}

// A fragment (16x32 f16): lane supplies row M=lane%16; element e holds
// K = k0 + 16*hl + e  (hl = lane/16)
__device__ __forceinline__ v16h load_a_frag(const _Float16* __restrict__ A,
                                            int row, int ldk, int k0, int hl) {
    const _Float16* p = A + (size_t)row * ldk + k0 + hl * 16;
    v8h lo = *(const v8h*)p;
    v8h hi = *(const v8h*)(p + 8);
    v16h a;
#pragma unroll
    for (int i = 0; i < 8; ++i) { a[i] = lo[i]; a[i + 8] = hi[i]; }
    return a;
}

// B fragment (32x16 f16): lane supplies column N=lane%16; element e holds
// K = k0 + 16*hl + e.  BT is stored [N][K] row-major -> contiguous v16h.
__device__ __forceinline__ v16h load_b_frag(const _Float16* __restrict__ BT,
                                            int col, int ldk, int k0, int hl) {
    return *(const v16h*)(BT + (size_t)col * ldk + k0 + hl * 16);
}

__device__ __forceinline__ void top5_insert(float d, int idx, float t[5], int ti[5]) {
    if (d < t[4]) {
        t[4] = d; ti[4] = idx;
#pragma unroll
        for (int j = 4; j > 0; --j) {
            if (t[j] < t[j - 1]) {
                float td = t[j]; t[j] = t[j - 1]; t[j - 1] = td;
                int   tx = ti[j]; ti[j] = ti[j - 1]; ti[j - 1] = tx;
            }
        }
    }
}

// ---------------- prep kernels ----------------

// keys f32 -> f16 and per-row squared norm. grid = NKEY, block = 256 (NC elems)
__global__ void __launch_bounds__(256)
prep_keys(const float* __restrict__ keys, _Float16* __restrict__ keys_h,
          float* __restrict__ k2) {
    __shared__ float red[256];
    int row = blockIdx.x, tid = threadIdx.x;
    float v = keys[(size_t)row * NC + tid];
    keys_h[(size_t)row * NC + tid] = (_Float16)v;
    red[tid] = v * v;
    __syncthreads();
    for (int off = 128; off > 0; off >>= 1) {
        if (tid < off) red[tid] += red[tid + off];
        __syncthreads();
    }
    if (tid == 0) k2[row] = red[0];
}

// W[K][N] f32 -> WT[N][K] f16
__global__ void __launch_bounds__(256)
transpose_to_f16(const float* __restrict__ W, _Float16* __restrict__ WT, int K, int N) {
    int idx = blockIdx.x * 256 + threadIdx.x;
    if (idx >= K * N) return;
    int kk = idx / N, nn = idx - kk * N;
    WT[(size_t)nn * K + kk] = (_Float16)W[idx];
}

__global__ void __launch_bounds__(256)
cvt_f16(const float* __restrict__ X, _Float16* __restrict__ Y, int n) {
    int i = blockIdx.x * 256 + threadIdx.x;
    if (i < n) Y[i] = (_Float16)X[i];
}

// ---------------- WMMA GEMM: out = act(A[M,K] @ W[K,N] + bias) ----------------
// W given as WT[N][K] f16.  Block = 256 thr = 8 waves, tile 64x64,
// each wave computes 16 rows x 32 cols (2 WMMAs / K-step).
__global__ void __launch_bounds__(256)
gemm_wmma(const _Float16* __restrict__ A, const _Float16* __restrict__ WT,
          const float* __restrict__ bias, float* __restrict__ out32,
          _Float16* __restrict__ out16, int M, int N, int K, int do_gelu) {
    int tid  = threadIdx.x;
    int lane = tid & 31;
    int wave = tid >> 5;
    int hl   = lane >> 4;
    int lm   = lane & 15;
    int m0 = blockIdx.y * 64 + (wave >> 1) * 16;
    int n0 = blockIdx.x * 64 + (wave & 1) * 32;
    v8f acc0 = {}; v8f acc1 = {};
    int rowA = m0 + lm;
    for (int k0 = 0; k0 < K; k0 += 32) {
        v16h a  = load_a_frag(A, rowA, K, k0, hl);
        v16h b0 = load_b_frag(WT, n0 + lm,      K, k0, hl);
        v16h b1 = load_b_frag(WT, n0 + 16 + lm, K, k0, hl);
        acc0 = __builtin_amdgcn_wmma_f32_16x16x32_f16(false, a, false, b0, (short)0, acc0, false, false);
        acc1 = __builtin_amdgcn_wmma_f32_16x16x32_f16(false, a, false, b1, (short)0, acc1, false, false);
    }
    int mb = m0 + hl * 8;
#pragma unroll
    for (int r = 0; r < 8; ++r) {
        int m = mb + r;
        {
            int n = n0 + lm;
            float v = acc0[r] + bias[n];
            if (do_gelu) v = gelu_exact(v);
            out32[(size_t)m * N + n] = v;
            if (out16) out16[(size_t)m * N + n] = (_Float16)v;
        }
        {
            int n = n0 + 16 + lm;
            float v = acc1[r] + bias[n];
            if (do_gelu) v = gelu_exact(v);
            out32[(size_t)m * N + n] = v;
            if (out16) out16[(size_t)m * N + n] = (_Float16)v;
        }
    }
}

// ---------------- LayerNorm (row-wise, biased var), f32 in -> f16 out ----------
__global__ void __launch_bounds__(256)
layernorm_f16(const float* __restrict__ X, const float* __restrict__ g,
              const float* __restrict__ b, _Float16* __restrict__ out, int N) {
    __shared__ float red[256], red2[256];
    int row = blockIdx.x, tid = threadIdx.x;
    const float* x = X + (size_t)row * N;
    float s = 0.f, s2 = 0.f;
    for (int i = tid; i < N; i += 256) { float v = x[i]; s += v; s2 += v * v; }
    red[tid] = s; red2[tid] = s2;
    __syncthreads();
    for (int off = 128; off > 0; off >>= 1) {
        if (tid < off) { red[tid] += red[tid + off]; red2[tid] += red2[tid + off]; }
        __syncthreads();
    }
    float mean = red[0] / N;
    float var  = red2[0] / N - mean * mean;
    float rstd = rsqrtf(var + 1e-5f);
    for (int i = tid; i < N; i += 256)
        out[(size_t)row * N + i] = (_Float16)(((x[i] - mean) * rstd) * g[i] + b[i]);
}

// ---------------- fused distance + partial top-5 ----------------
// grid = (SLICES, NB/QT), block = 128 (4 waves).  Block owns QT=32 queries and
// a key slice; each wave streams 16 keys/iter against both 16-query column
// tiles (2 WMMAs, A fragment reused -> halves streaming L2 traffic vs QT=16).
// d = |q|^2 + |k|^2 - 2 q.k ; per-lane register top-5 per column tile, LDS
// merge, per-slice partial result to global.
__global__ void __launch_bounds__(128)
dist_topk_partial(const _Float16* __restrict__ keys_h, const _Float16* __restrict__ qc_h,
                  const float* __restrict__ qc_f32, const float* __restrict__ k2,
                  float* __restrict__ part_d, int* __restrict__ part_i) {
    __shared__ float q2s[QT];
    __shared__ float cd[QT][8][5];
    __shared__ int   ci[QT][8][5];
    int tid = threadIdx.x, lane = tid & 31, wave = tid >> 5;
    int hl = lane >> 4, lm = lane & 15;
    int q0 = blockIdx.y * QT;
    int ks = blockIdx.x * KEYS_PER_SLICE;
    int ke = ks + KEYS_PER_SLICE;

    if (tid < QT) q2s[tid] = 0.f;
    __syncthreads();
    {   // |q|^2 for the QT queries: 4 thread-groups x 64 cols each
        int q = tid & (QT - 1), gidx = tid / QT;
        const float* qr = qc_f32 + (size_t)(q0 + q) * NC + gidx * 64;
        float s = 0.f;
#pragma unroll
        for (int c = 0; c < 64; ++c) { float v = qr[c]; s += v * v; }
        atomicAdd(&q2s[q], s);
    }
    __syncthreads();

    float t0[5], t1[5]; int ti0[5], ti1[5];
#pragma unroll
    for (int j = 0; j < 5; ++j) {
        t0[j] = 3.4e38f; ti0[j] = 0;
        t1[j] = 3.4e38f; ti1[j] = 0;
    }
    float q2v0 = q2s[lm];
    float q2v1 = q2s[16 + lm];

    int iters = (KEYS_PER_SLICE + 63) / 64;
    for (int it = 0; it < iters; ++it) {
        int kb = ks + it * 64 + wave * 16;
        int rowA = kb + lm; if (rowA >= NKEY) rowA = NKEY - 1;
        // prefetch next iteration's key row into the cache hierarchy
        {
            int pref = rowA + 64;
            if (pref < NKEY) __builtin_prefetch(keys_h + (size_t)pref * NC, 0, 1);
        }
        v8f acc0 = {}; v8f acc1 = {};
#pragma unroll
        for (int kk = 0; kk < 8; ++kk) {
            int k0 = kk * 32;
            v16h a  = load_a_frag(keys_h, rowA, NC, k0, hl);
            v16h b0 = load_b_frag(qc_h, q0 + lm,      NC, k0, hl);   // L0-hot
            v16h b1 = load_b_frag(qc_h, q0 + 16 + lm, NC, k0, hl);   // L0-hot
            acc0 = __builtin_amdgcn_wmma_f32_16x16x32_f16(false, a, false, b0, (short)0, acc0, false, false);
            acc1 = __builtin_amdgcn_wmma_f32_16x16x32_f16(false, a, false, b1, (short)0, acc1, false, false);
        }
        int keyb = kb + hl * 8;
#pragma unroll
        for (int r = 0; r < 8; ++r) {
            int key = keyb + r;
            if (key < ke) {
                float kk2 = k2[key];
                float d0 = fmaxf(q2v0 + kk2 - 2.0f * acc0[r], 0.f);
                float d1 = fmaxf(q2v1 + kk2 - 2.0f * acc1[r], 0.f);
                top5_insert(d0, key, t0, ti0);
                top5_insert(d1, key, t1, ti1);
            }
        }
    }

    int slot = wave * 2 + hl;   // 8 unique (wave, half) streams per query
#pragma unroll
    for (int j = 0; j < 5; ++j) {
        cd[lm][slot][j]      = t0[j]; ci[lm][slot][j]      = ti0[j];
        cd[16 + lm][slot][j] = t1[j]; ci[16 + lm][slot][j] = ti1[j];
    }
    __syncthreads();
    if (tid < QT) {
        float m[5]; int mi[5];
#pragma unroll
        for (int j = 0; j < 5; ++j) { m[j] = 3.4e38f; mi[j] = 0; }
        for (int s = 0; s < 8; ++s)
            for (int j = 0; j < 5; ++j) top5_insert(cd[tid][s][j], ci[tid][s][j], m, mi);
        float* pd = part_d + ((size_t)(q0 + tid) * SLICES + blockIdx.x) * TOPK;
        int*   pi = part_i + ((size_t)(q0 + tid) * SLICES + blockIdx.x) * TOPK;
        for (int j = 0; j < 5; ++j) { pd[j] = m[j]; pi[j] = mi[j]; }
    }
}

// merge 8 slice-partials -> global top-5, weights, confidence
__global__ void __launch_bounds__(64)
topk_finalize(const float* __restrict__ part_d, const int* __restrict__ part_i,
              float* __restrict__ w_out, int* __restrict__ i_out,
              float* __restrict__ conf) {
    int q = blockIdx.x * 64 + threadIdx.x;
    if (q >= NB) return;
    float t[5]; int ti[5];
#pragma unroll
    for (int j = 0; j < 5; ++j) { t[j] = 3.4e38f; ti[j] = 0; }
    const float* pd = part_d + (size_t)q * SLICES * TOPK;
    const int*   pi = part_i + (size_t)q * SLICES * TOPK;
    for (int s = 0; s < SLICES * TOPK; ++s) top5_insert(pd[s], pi[s], t, ti);
    float w[5], sum = 0.f;
#pragma unroll
    for (int j = 0; j < 5; ++j) { w[j] = 1.f / (t[j] + 1e-6f); sum += w[j]; }
#pragma unroll
    for (int j = 0; j < 5; ++j) { w_out[q * TOPK + j] = w[j] / sum; i_out[q * TOPK + j] = ti[j]; }
    conf[q] = 1.f / (t[0] + 1e-6f);
}

// gather values_c rows by top-k index -> f16 matrix [NB*TOPK, NC]
__global__ void __launch_bounds__(256)
gather_values(const float* __restrict__ values, const int* __restrict__ idx,
              _Float16* __restrict__ V) {
    int r = blockIdx.x;           // 0 .. NB*TOPK-1
    int t = threadIdx.x;          // 0 .. 255 (= NC)
    int id = idx[r];
    V[(size_t)r * NC + t] = (_Float16)values[(size_t)id * NC + t];
}

// retrieved[b,:] = sum_j w[b,j] * dec[b*5+j,:]
__global__ void __launch_bounds__(256)
weighted_sum(const float* __restrict__ dec, const float* __restrict__ w,
             float* __restrict__ out) {
    int b = blockIdx.x, t = threadIdx.x;
    float wl[5];
#pragma unroll
    for (int j = 0; j < 5; ++j) wl[j] = w[b * TOPK + j];
    for (int d = t; d < ND; d += 256) {
        float s = 0.f;
#pragma unroll
        for (int j = 0; j < 5; ++j)
            s += wl[j] * dec[((size_t)b * TOPK + j) * ND + d];
        out[(size_t)b * ND + d] = s;
    }
}

// ---------------- host ----------------

extern "C" void kernel_launch(void* const* d_in, const int* in_sizes, int n_in,
                              void* d_out, int out_size, void* d_ws, size_t ws_size,
                              hipStream_t stream) {
    const float* query    = (const float*)d_in[0];
    const float* keys_c   = (const float*)d_in[1];
    const float* values_c = (const float*)d_in[2];
    const float* cW1 = (const float*)d_in[3];  const float* cb1 = (const float*)d_in[4];
    const float* cg1 = (const float*)d_in[5];  const float* cB1 = (const float*)d_in[6];
    const float* cW2 = (const float*)d_in[7];  const float* cb2 = (const float*)d_in[8];
    const float* cg2 = (const float*)d_in[9];  const float* cB2 = (const float*)d_in[10];
    const float* cW3 = (const float*)d_in[11]; const float* cb3 = (const float*)d_in[12];
    const float* dW1 = (const float*)d_in[13]; const float* db1 = (const float*)d_in[14];
    const float* dg1 = (const float*)d_in[15]; const float* dB1 = (const float*)d_in[16];
    const float* dW2 = (const float*)d_in[17]; const float* db2 = (const float*)d_in[18];
    const float* dg2 = (const float*)d_in[19]; const float* dB2 = (const float*)d_in[20];
    const float* dW3 = (const float*)d_in[21]; const float* db3 = (const float*)d_in[22];

    float* retrieved = (float*)d_out;                   // [NB, ND]
    float* conf      = (float*)d_out + (size_t)NB * ND; // [NB]

    // workspace carve-out
    char* p = (char*)d_ws;
    auto alloc = [&](size_t bytes) -> char* {
        char* r = p; p += (bytes + 255) & ~(size_t)255; return r;
    };
    _Float16* q_h    = (_Float16*)alloc((size_t)NB * ND * 2);
    _Float16* keys_h = (_Float16*)alloc((size_t)NKEY * NC * 2);
    float*    k2     = (float*)   alloc((size_t)NKEY * 4);
    _Float16* cW1T = (_Float16*)alloc((size_t)512  * 1024 * 2);
    _Float16* cW2T = (_Float16*)alloc((size_t)256  * 512  * 2);
    _Float16* cW3T = (_Float16*)alloc((size_t)256  * 256  * 2);
    _Float16* dW1T = (_Float16*)alloc((size_t)256  * 256  * 2);
    _Float16* dW2T = (_Float16*)alloc((size_t)512  * 256  * 2);
    _Float16* dW3T = (_Float16*)alloc((size_t)1024 * 512  * 2);
    float*    h1_f = (float*)   alloc((size_t)NB * 512 * 4);
    _Float16* h1_h = (_Float16*)alloc((size_t)NB * 512 * 2);
    float*    h2_f = (float*)   alloc((size_t)NB * 256 * 4);
    _Float16* h2_h = (_Float16*)alloc((size_t)NB * 256 * 2);
    float*    qc_f = (float*)   alloc((size_t)NB * NC * 4);
    _Float16* qc_h = (_Float16*)alloc((size_t)NB * NC * 2);
    float*    part_d = (float*) alloc((size_t)NB * SLICES * TOPK * 4);
    int*      part_i = (int*)   alloc((size_t)NB * SLICES * TOPK * 4);
    float*    tk_w   = (float*) alloc((size_t)NB * TOPK * 4);
    int*      tk_i   = (int*)   alloc((size_t)NB * TOPK * 4);
    _Float16* V_h  = (_Float16*)alloc((size_t)NB * TOPK * NC * 2);
    float*    g1_f = (float*)   alloc((size_t)NB * TOPK * 256 * 4);
    _Float16* g1_h = (_Float16*)alloc((size_t)NB * TOPK * 256 * 2);
    float*    g2_f = (float*)   alloc((size_t)NB * TOPK * 512 * 4);
    _Float16* g2_h = (_Float16*)alloc((size_t)NB * TOPK * 512 * 2);
    float*    dec_f = (float*)  alloc((size_t)NB * TOPK * ND * 4);
    (void)ws_size; (void)in_sizes; (void)n_in; (void)out_size;

    const int MR = NB * TOPK; // 5120 decompress rows

    // --- preprocessing ---
    hipLaunchKernelGGL(prep_keys, dim3(NKEY), dim3(256), 0, stream, keys_c, keys_h, k2);
    hipLaunchKernelGGL(cvt_f16, dim3((NB * ND + 255) / 256), dim3(256), 0, stream,
                       query, q_h, NB * ND);
    hipLaunchKernelGGL(transpose_to_f16, dim3((1024 * 512 + 255) / 256), dim3(256), 0, stream, cW1, cW1T, 1024, 512);
    hipLaunchKernelGGL(transpose_to_f16, dim3((512 * 256 + 255) / 256),  dim3(256), 0, stream, cW2, cW2T, 512, 256);
    hipLaunchKernelGGL(transpose_to_f16, dim3((256 * 256 + 255) / 256),  dim3(256), 0, stream, cW3, cW3T, 256, 256);
    hipLaunchKernelGGL(transpose_to_f16, dim3((256 * 256 + 255) / 256),  dim3(256), 0, stream, dW1, dW1T, 256, 256);
    hipLaunchKernelGGL(transpose_to_f16, dim3((256 * 512 + 255) / 256),  dim3(256), 0, stream, dW2, dW2T, 256, 512);
    hipLaunchKernelGGL(transpose_to_f16, dim3((512 * 1024 + 255) / 256), dim3(256), 0, stream, dW3, dW3T, 512, 1024);

    // --- compress MLP: 1024 -> 512 -> 256 -> 256 ---
    hipLaunchKernelGGL(gemm_wmma, dim3(512 / 64, NB / 64), dim3(256), 0, stream,
                       q_h, cW1T, cb1, h1_f, (_Float16*)nullptr, NB, 512, 1024, 1);
    hipLaunchKernelGGL(layernorm_f16, dim3(NB), dim3(256), 0, stream, h1_f, cg1, cB1, h1_h, 512);
    hipLaunchKernelGGL(gemm_wmma, dim3(256 / 64, NB / 64), dim3(256), 0, stream,
                       h1_h, cW2T, cb2, h2_f, (_Float16*)nullptr, NB, 256, 512, 1);
    hipLaunchKernelGGL(layernorm_f16, dim3(NB), dim3(256), 0, stream, h2_f, cg2, cB2, h2_h, 256);
    hipLaunchKernelGGL(gemm_wmma, dim3(256 / 64, NB / 64), dim3(256), 0, stream,
                       h2_h, cW3T, cb3, qc_f, qc_h, NB, 256, 256, 0);

    // --- fused distance + top-5 (dominant WMMA work, keys resident in L2) ---
    hipLaunchKernelGGL(dist_topk_partial, dim3(SLICES, NB / QT), dim3(128), 0, stream,
                       keys_h, qc_h, qc_f, k2, part_d, part_i);
    hipLaunchKernelGGL(topk_finalize, dim3(NB / 64), dim3(64), 0, stream,
                       part_d, part_i, tk_w, tk_i, conf);

    // --- gather + decompress MLP: 256 -> 256 -> 512 -> 1024 on 5120 rows ---
    hipLaunchKernelGGL(gather_values, dim3(MR), dim3(256), 0, stream, values_c, tk_i, V_h);
    hipLaunchKernelGGL(gemm_wmma, dim3(256 / 64, MR / 64), dim3(256), 0, stream,
                       V_h, dW1T, db1, g1_f, (_Float16*)nullptr, MR, 256, 256, 1);
    hipLaunchKernelGGL(layernorm_f16, dim3(MR), dim3(256), 0, stream, g1_f, dg1, dB1, g1_h, 256);
    hipLaunchKernelGGL(gemm_wmma, dim3(512 / 64, MR / 64), dim3(256), 0, stream,
                       g1_h, dW2T, db2, g2_f, (_Float16*)nullptr, MR, 512, 256, 1);
    hipLaunchKernelGGL(layernorm_f16, dim3(MR), dim3(256), 0, stream, g2_f, dg2, dB2, g2_h, 512);
    hipLaunchKernelGGL(gemm_wmma, dim3(ND / 64, MR / 64), dim3(256), 0, stream,
                       g2_h, dW3T, db3, dec_f, (_Float16*)nullptr, MR, ND, 512, 0);

    // --- inverse-distance weighted combine ---
    hipLaunchKernelGGL(weighted_sum, dim3(NB), dim3(256), 0, stream, dec_f, tk_w, retrieved);
}
